// PrefixCausalAttention_90598040142117
// MI455X (gfx1250) — matrix operand order repaired and checked by reading, not compiled, not run on hardware
//
#include <hip/hip_runtime.h>
#include <hip/hip_bf16.h>

// ---------------------------------------------------------------------------
// MI455X (gfx1250) prefix-causal attention, bf16 WMMA pipeline.
//   pre-pass converts (x -> bf16, w -> transposed bf16), then:
//   qkv GEMM -> flash attention (online softmax) -> proj GEMM (+bias)
// All matrix math: v_wmma_f32_16x16x32_bf16 (wave32).
// Hot GEMM loops: async global->LDS staging (ASYNCcnt) + WMMA, no ALU.
// ---------------------------------------------------------------------------

typedef __attribute__((ext_vector_type(16))) __bf16 v16bf;
typedef __attribute__((ext_vector_type(8)))  float  v8f;

// --- CDNA5 async global->LDS path (guarded; falls back to sync copies) ---
#if defined(__has_builtin)
#  if __has_builtin(__builtin_amdgcn_global_load_async_to_lds_b128)
#    define ASYNC_LDS 1
#  endif
#endif

#ifdef ASYNC_LDS
typedef int v4i __attribute__((vector_size(16)));
typedef __attribute__((address_space(1))) v4i* gv4_ptr;
typedef __attribute__((address_space(3))) v4i* lv4_ptr;
#  define ASYNC_CP16(dst, src)                                                  \
      __builtin_amdgcn_global_load_async_to_lds_b128(                           \
          (gv4_ptr)(src), (lv4_ptr)(dst), 0, 0)
#  if __has_builtin(__builtin_amdgcn_s_wait_asynccnt)
#    define WAIT_ASYNC() __builtin_amdgcn_s_wait_asynccnt(0)
#  else
#    define WAIT_ASYNC() asm volatile("s_wait_asynccnt 0x0" ::: "memory")
#  endif
#endif

union Frag {
    v16bf v;
    uint4 q[2];
    unsigned short s[16];
};

union Pack8 {
    unsigned short s[8];
    uint4 q;
};

union Pack4 {
    unsigned short s[4];
    uint2 q;
};

static __device__ __forceinline__ unsigned short f2bf(float f) {
    unsigned int u = __builtin_bit_cast(unsigned int, f);
    u += 0x7fffu + ((u >> 16) & 1u);            // round-to-nearest-even
    return (unsigned short)(u >> 16);
}

static __device__ __forceinline__ v8f wmma_bf16(const v16bf& a, const v16bf& b, const v8f& c) {
    return __builtin_amdgcn_wmma_f32_16x16x32_bf16(false, a, false, b, (short)0, c, false, false);
}

#define NHEADS 16
#define DHEAD  64
#define NSEQ   256
#define CCH    1024
#define NEGF   (-3.0e38f)

// ---------------------------------------------------------------------------
// Pre-pass A: elementwise f32 -> bf16 (vectorized, 8 elems/thread).
// ---------------------------------------------------------------------------
__global__ __launch_bounds__(256) void cvt_bf16_kernel(
    const float* __restrict__ in, unsigned short* __restrict__ out)
{
    const size_t i = ((size_t)blockIdx.x * 256 + threadIdx.x) * 8;
    float4 a = *(const float4*)(in + i);
    float4 b = *(const float4*)(in + i + 4);
    Pack8 p;
    p.s[0] = f2bf(a.x); p.s[1] = f2bf(a.y); p.s[2] = f2bf(a.z); p.s[3] = f2bf(a.w);
    p.s[4] = f2bf(b.x); p.s[5] = f2bf(b.y); p.s[6] = f2bf(b.z); p.s[7] = f2bf(b.w);
    *(uint4*)(out + i) = p.q;
}

// ---------------------------------------------------------------------------
// Pre-pass B: w[k][n] f32 -> wT[n][k] bf16 via 32x32 LDS tile.  K is always
// CCH (=1024); N passed in (3072 for w_qkv, 1024 for w_proj).
// ---------------------------------------------------------------------------
__global__ __launch_bounds__(256) void transpose_cvt_kernel(
    const float* __restrict__ w, unsigned short* __restrict__ wT, int N)
{
    __shared__ __align__(16) unsigned short T[32][36];
    const int n0 = blockIdx.x * 32;
    const int k0 = blockIdx.y * 32;

    const int kk = threadIdx.x >> 3;          // 0..31
    const int nn = (threadIdx.x & 7) * 4;     // 0..28
    float4 f = *(const float4*)(w + (size_t)(k0 + kk) * N + n0 + nn);
    T[kk][nn + 0] = f2bf(f.x);
    T[kk][nn + 1] = f2bf(f.y);
    T[kk][nn + 2] = f2bf(f.z);
    T[kk][nn + 3] = f2bf(f.w);
    __syncthreads();

    const int nn2 = threadIdx.x >> 3;         // 0..31
    const int kk2 = (threadIdx.x & 7) * 4;    // 0..28
    Pack4 p;
#pragma unroll
    for (int j = 0; j < 4; ++j) p.s[j] = T[kk2 + j][nn2];
    *(uint2*)(&wT[(size_t)(n0 + nn2) * CCH + k0 + kk2]) = p.q;
}

// ---------------------------------------------------------------------------
// Shared GEMM core: C[128x128] = A_bf16[M x 1024] @ wT_bf16[N x 1024]^T
// 8 waves as 4(M) x 2(N); wave tile 32x64 = 2x4 C-frags; BK=32.
// ---------------------------------------------------------------------------
#ifdef ASYNC_LDS
#define GEMM_STAGE(sa, sb)                                                       \
            ASYNC_CP16(&At[srow][scc],     sa);                                  \
            ASYNC_CP16(&At[srow][scc + 8], sa + 8);                              \
            ASYNC_CP16(&Bt[srow][scc],     sb);                                  \
            ASYNC_CP16(&Bt[srow][scc + 8], sb + 8);                              \
            WAIT_ASYNC();
#else
#define GEMM_STAGE(sa, sb)                                                       \
            *(uint4*)(&At[srow][scc])     = *(const uint4*)(sa);                 \
            *(uint4*)(&At[srow][scc + 8]) = *(const uint4*)(sa + 8);             \
            *(uint4*)(&Bt[srow][scc])     = *(const uint4*)(sb);                 \
            *(uint4*)(&Bt[srow][scc + 8]) = *(const uint4*)(sb + 8);
#endif

#define GEMM_CORE(A_PTR, BT_PTR)                                                 \
    __shared__ __align__(16) unsigned short At[128][40];                         \
    __shared__ __align__(16) unsigned short Bt[128][40];                         \
    const int lane  = threadIdx.x & 31;                                          \
    const int wv    = threadIdx.x >> 5;                                          \
    const int wm    = wv >> 1;                                                   \
    const int wn    = wv & 1;                                                    \
    const int m0    = blockIdx.x * 128;                                          \
    const int n0    = blockIdx.y * 128;                                          \
    const int lrow  = lane & 15;                                                 \
    const bool hi   = lane >= 16;                                                \
    const int rbase = hi ? 8 : 0;                                                \
    v8f acc[2][4];                                                               \
    _Pragma("unroll")                                                            \
    for (int i = 0; i < 2; ++i)                                                  \
        _Pragma("unroll")                                                        \
        for (int j = 0; j < 4; ++j)                                              \
            _Pragma("unroll")                                                    \
            for (int r = 0; r < 8; ++r) acc[i][j][r] = 0.0f;                     \
    const int srow = threadIdx.x >> 1;                                           \
    const int scc  = (threadIdx.x & 1) * 16;                                     \
    for (int k0 = 0; k0 < CCH; k0 += 32) {                                       \
        __syncthreads();                                                         \
        {                                                                        \
            const unsigned short* sa = (A_PTR) + (size_t)(m0 + srow) * CCH + k0 + scc; \
            const unsigned short* sb = (BT_PTR) + (size_t)(n0 + srow) * CCH + k0 + scc; \
            if (k0 + 32 < CCH) {                                                 \
                __builtin_prefetch(sa + 32, 0, 3);                               \
                __builtin_prefetch(sb + 32, 0, 3);                               \
            }                                                                    \
            GEMM_STAGE(sa, sb)                                                   \
        }                                                                        \
        __syncthreads();                                                         \
        Frag af[2], bfr[4];                                                      \
        _Pragma("unroll")                                                        \
        for (int fm = 0; fm < 2; ++fm) {                                         \
            const int row = wm * 32 + fm * 16 + lrow;                            \
            const int klo = hi ? 8 : 0;                                          \
            af[fm].q[0] = *(const uint4*)(&At[row][klo]);                        \
            af[fm].q[1] = *(const uint4*)(&At[row][klo + 16]);                   \
        }                                                                        \
        _Pragma("unroll")                                                        \
        for (int fn = 0; fn < 4; ++fn) {                                         \
            const int row  = wn * 64 + fn * 16 + lrow;                           \
            const int koff = hi ? 8 : 0;                                         \
            bfr[fn].q[0] = *(const uint4*)(&Bt[row][koff]);                      \
            bfr[fn].q[1] = *(const uint4*)(&Bt[row][koff + 16]);                 \
        }                                                                        \
        _Pragma("unroll")                                                        \
        for (int fm = 0; fm < 2; ++fm)                                           \
            _Pragma("unroll")                                                    \
            for (int fn = 0; fn < 4; ++fn)                                       \
                acc[fm][fn] = wmma_bf16(af[fm].v, bfr[fn].v, acc[fm][fn]);       \
    }

// ---------------------------------------------------------------------------
// Kernel 1: qkv GEMM.  Epilogue scatters bf16 into per-head q/k/v buffers.
// Block cols span 128 = 2 heads, never crossing the s (q/k/v) boundary.
// ---------------------------------------------------------------------------
__global__ __launch_bounds__(256) void qkv_gemm_kernel(
    const unsigned short* __restrict__ x_bf, const unsigned short* __restrict__ wT,
    unsigned short* __restrict__ qb, unsigned short* __restrict__ kb,
    unsigned short* __restrict__ vb)
{
    GEMM_CORE(x_bf, wT)

    const int sIdx = n0 >> 10;
    unsigned short* base = (sIdx == 0) ? qb : ((sIdx == 1) ? kb : vb);
#pragma unroll
    for (int fm = 0; fm < 2; ++fm)
#pragma unroll
        for (int fn = 0; fn < 4; ++fn) {
            const int c1 = (n0 + wn * 64 + fn * 16 + lrow) & 1023;
            const int hh = c1 >> 6;
            const int dd = c1 & 63;
#pragma unroll
            for (int r = 0; r < 8; ++r) {
                const int rowM = m0 + wm * 32 + fm * 16 + rbase + r;
                const int bIdx = rowM >> 8;
                const int n    = rowM & 255;
                base[((size_t)(bIdx * NHEADS + hh) * NSEQ + n) * DHEAD + dd] =
                    f2bf(acc[fm][fn][r]);
            }
        }
}

// ---------------------------------------------------------------------------
// Kernel 3: proj GEMM.  Epilogue adds bias, writes f32.
// ---------------------------------------------------------------------------
__global__ __launch_bounds__(256) void proj_gemm_kernel(
    const unsigned short* __restrict__ a_bf, const unsigned short* __restrict__ wT,
    const float* __restrict__ bias, float* __restrict__ out)
{
    GEMM_CORE(a_bf, wT)

#pragma unroll
    for (int fm = 0; fm < 2; ++fm)
#pragma unroll
        for (int fn = 0; fn < 4; ++fn) {
            const int col = n0 + wn * 64 + fn * 16 + lrow;
            const float bv = bias[col];
#pragma unroll
            for (int r = 0; r < 8; ++r) {
                const int rowM = m0 + wm * 32 + fm * 16 + rbase + r;
                out[(size_t)rowM * CCH + col] = acc[fm][fn][r] + bv;
            }
        }
}

// ---------------------------------------------------------------------------
// Kernel 2: flash attention.  One block per (b, h, half).  8 waves, each wave
// owns 16 query rows; stream K/V in 32-key chunks through LDS with online
// softmax.  V staged transposed so P@V B-frags are contiguous per lane.
// ---------------------------------------------------------------------------
__global__ __launch_bounds__(256) void flash_attn_kernel(
    const unsigned short* __restrict__ qb, const unsigned short* __restrict__ kb,
    const unsigned short* __restrict__ vb, const unsigned char* __restrict__ mask,
    unsigned short* __restrict__ attn_out)
{
    __shared__ __align__(16) unsigned short Ks[32][72];       // [key][d], padded
    __shared__ __align__(16) unsigned short Vt[64][40];       // [d][key], padded
    __shared__ __align__(16) unsigned short Ps[8][16][40];    // per-wave P staging

    const int lane  = threadIdx.x & 31;
    const int wv    = threadIdx.x >> 5;
    const int bh    = blockIdx.x >> 1;
    const int half  = blockIdx.x & 1;
    const int b     = bh >> 4;
    const int h     = bh & 15;
    const int q0    = half * 128 + wv * 16;
    const int lrow  = lane & 15;
    const bool hi   = lane >= 16;
    const int rbase = hi ? 8 : 0;

    const size_t headBase = (size_t)bh * NSEQ * DHEAD;

    // Q A-fragments (16 rows x 64 kdim = 2 frags), loaded once from global.
    Frag qa[2];
    {
        const unsigned short* qrow = qb + headBase + (size_t)(q0 + lrow) * DHEAD;
#pragma unroll
        for (int f = 0; f < 2; ++f) {
            const int klo = (hi ? 8 : 0) + f * 32;
            qa[f].q[0] = *(const uint4*)(qrow + klo);
            qa[f].q[1] = *(const uint4*)(qrow + klo + 16);
        }
    }

    v8f o[4];
#pragma unroll
    for (int i = 0; i < 4; ++i)
#pragma unroll
        for (int r = 0; r < 8; ++r) o[i][r] = 0.0f;
    float mrun[8], lrun[8];
#pragma unroll
    for (int r = 0; r < 8; ++r) { mrun[r] = NEGF; lrun[r] = 0.0f; }

    const int ldKey = threadIdx.x >> 3;        // 0..31
    const int ldDc  = (threadIdx.x & 7) * 8;   // 0..56

    for (int kb0 = 0; kb0 < NSEQ; kb0 += 32) {
        __syncthreads();
        // --- stage K chunk row-major (async), V chunk transposed ---
        {
            const unsigned short* krow = kb + headBase + (size_t)(kb0 + ldKey) * DHEAD + ldDc;
#ifdef ASYNC_LDS
            ASYNC_CP16(&Ks[ldKey][ldDc], krow);
#else
            *(uint4*)(&Ks[ldKey][ldDc]) = *(const uint4*)krow;
#endif
            const unsigned short* vrow = vb + headBase + (size_t)(kb0 + ldKey) * DHEAD + ldDc;
            Pack8 vv; vv.q = *(const uint4*)vrow;
#pragma unroll
            for (int j = 0; j < 8; ++j) Vt[ldDc + j][ldKey] = vv.s[j];
            if (kb0 + 32 < NSEQ) __builtin_prefetch(krow + 32 * DHEAD, 0, 3);
#ifdef ASYNC_LDS
            WAIT_ASYNC();
#endif
        }
        __syncthreads();

        // --- S = Q K^T  (two 16-col groups) ---
        v8f sfr[2];
#pragma unroll
        for (int g = 0; g < 2; ++g) {
            v8f a;
#pragma unroll
            for (int r = 0; r < 8; ++r) a[r] = 0.0f;
#pragma unroll
            for (int f = 0; f < 2; ++f) {
                Frag kf;
                const int key = g * 16 + lrow;
                const int ddb = f * 32 + (hi ? 8 : 0);
                kf.q[0] = *(const uint4*)(&Ks[key][ddb]);
                kf.q[1] = *(const uint4*)(&Ks[key][ddb + 16]);
                a = wmma_bf16(qa[f].v, kf.v, a);
            }
            sfr[g] = a;
        }

        // --- scale + prefix-causal mask ---
#pragma unroll
        for (int g = 0; g < 2; ++g) {
            const int kcol = kb0 + g * 16 + lrow;
            const bool mk  = mask[b * NSEQ + kcol] != 0;
#pragma unroll
            for (int r = 0; r < 8; ++r) {
                const int qrow = q0 + rbase + r;
                const bool allow = (qrow >= kcol) || (!mk);
                sfr[g][r] = allow ? sfr[g][r] * 0.125f : NEGF;
            }
        }

        // --- online softmax (rows live in VGPR index; cols across 16 lanes) ---
        float bm[8], p0[8], p1[8], rs[8], alpha[8];
#pragma unroll
        for (int r = 0; r < 8; ++r) bm[r] = fmaxf(sfr[0][r], sfr[1][r]);
        for (int off = 1; off <= 8; off <<= 1)
#pragma unroll
            for (int r = 0; r < 8; ++r) bm[r] = fmaxf(bm[r], __shfl_xor(bm[r], off, 32));
#pragma unroll
        for (int r = 0; r < 8; ++r) {
            const float mn = fmaxf(mrun[r], bm[r]);
            alpha[r] = __expf(mrun[r] - mn);
            p0[r] = __expf(sfr[0][r] - mn);
            p1[r] = __expf(sfr[1][r] - mn);
            rs[r] = p0[r] + p1[r];
            mrun[r] = mn;
            lrun[r] *= alpha[r];
        }
        for (int off = 1; off <= 8; off <<= 1)
#pragma unroll
            for (int r = 0; r < 8; ++r) rs[r] += __shfl_xor(rs[r], off, 32);
#pragma unroll
        for (int r = 0; r < 8; ++r) lrun[r] += rs[r];
#pragma unroll
        for (int i = 0; i < 4; ++i)
#pragma unroll
            for (int r = 0; r < 8; ++r) o[i][r] *= alpha[r];

        // --- P: C-layout -> A-layout via per-wave LDS staging ---
#pragma unroll
        for (int r = 0; r < 8; ++r) {
            Ps[wv][rbase + r][lrow]      = f2bf(p0[r]);
            Ps[wv][rbase + r][16 + lrow] = f2bf(p1[r]);
        }
        Frag pa;
        {
            const int klo = hi ? 8 : 0;
            pa.q[0] = *(const uint4*)(&Ps[wv][lrow][klo]);
            pa.q[1] = *(const uint4*)(&Ps[wv][lrow][klo + 16]);
        }

        // --- O += P @ V ---
#pragma unroll
        for (int dchunk = 0; dchunk < 4; ++dchunk) {
            Frag vf;
            const int drow = dchunk * 16 + lrow;
            const int koff = hi ? 8 : 0;
            vf.q[0] = *(const uint4*)(&Vt[drow][koff]);
            vf.q[1] = *(const uint4*)(&Vt[drow][koff + 16]);
            o[dchunk] = wmma_bf16(pa.v, vf.v, o[dchunk]);
        }
    }

    // --- normalize and write bf16 [b][n][h*64+d] ---
#pragma unroll
    for (int dchunk = 0; dchunk < 4; ++dchunk)
#pragma unroll
        for (int r = 0; r < 8; ++r) {
            const int qrow = q0 + rbase + r;
            const float val = o[dchunk][r] / lrun[r];
            attn_out[(size_t)(b * NSEQ + qrow) * CCH + h * DHEAD + dchunk * 16 + lrow] =
                f2bf(val);
        }
}

// ---------------------------------------------------------------------------
extern "C" void kernel_launch(void* const* d_in, const int* in_sizes, int n_in,
                              void* d_out, int out_size, void* d_ws, size_t ws_size,
                              hipStream_t stream) {
    const float*         x      = (const float*)d_in[0];
    const unsigned char* mask   = (const unsigned char*)d_in[1];
    const float*         w_qkv  = (const float*)d_in[2];
    const float*         w_proj = (const float*)d_in[3];
    const float*         b_proj = (const float*)d_in[4];
    float*               out    = (float*)d_out;

    const size_t headElems = (size_t)64 * NHEADS * NSEQ * DHEAD;  // 16,777,216
    unsigned short* qb     = (unsigned short*)d_ws;
    unsigned short* kbuf   = qb + headElems;
    unsigned short* vbuf   = kbuf + headElems;
    unsigned short* attn   = vbuf + headElems;
    unsigned short* x_bf   = attn + headElems;                    // 16,777,216 elems
    unsigned short* wqkvT  = x_bf + headElems;                    // 3,145,728 elems
    unsigned short* wprojT = wqkvT + (size_t)CCH * 3 * CCH;       // 1,048,576 elems

    // --- pre-passes: convert / transpose-convert (memory-bound) ---
    cvt_bf16_kernel<<<8192, 256, 0, stream>>>(x, x_bf);
    transpose_cvt_kernel<<<dim3(96, 32), 256, 0, stream>>>(w_qkv, wqkvT, 3 * CCH);
    transpose_cvt_kernel<<<dim3(32, 32), 256, 0, stream>>>(w_proj, wprojT, CCH);

    // --- qkv: M=16384 (128 tiles), N=3072 (24 tiles of 128) ---
    qkv_gemm_kernel<<<dim3(128, 24), 256, 0, stream>>>(x_bf, wqkvT, qb, kbuf, vbuf);
    // --- attention: B*H*2 = 2048 blocks ---
    flash_attn_kernel<<<2048, 256, 0, stream>>>(qb, kbuf, vbuf, mask, attn);
    // --- proj: M=16384 (128 tiles), N=1024 (8 tiles of 128) ---
    proj_gemm_kernel<<<dim3(128, 8), 256, 0, stream>>>(attn, wprojT, b_proj, out);
}
